// SCLIPLayers_25967372271997
// MI455X (gfx1250) — compile-verified, matrix operand order checked by reading, not run on hardware
//
#include <hip/hip_runtime.h>
#include <hip/hip_bf16.h>

// ---------------------------------------------------------------------------
// SCLIP 2-layer block for MI455X (gfx1250): all GEMMs via v_wmma_f32_16x16x32_bf16
// ---------------------------------------------------------------------------

typedef unsigned short u16;
typedef __attribute__((ext_vector_type(16))) __bf16 bf16x16;
typedef __attribute__((ext_vector_type(8)))  float  f32x8;

#define LQ   577
#define BQ   8
#define CQ   1024
#define HQ   16
#define DQ   64
#define DFF  4096
#define LB   (LQ*BQ)      // 4616 token rows
#define LP   608          // L padded to multiple of 32 (K dim of attn-apply)
#define LBP  (LP*BQ)      // 4864 padded token rows (covers 64-row M tiles)
#define ATTR 640          // attn rows padded to multiple of 64
#define NLAY 2

__device__ __forceinline__ u16 f2bf(float f) {
  unsigned u = __float_as_uint(f);
  u += 0x7fffu + ((u >> 16) & 1u);          // round-to-nearest-even
  return (u16)(u >> 16);
}

// Load one 16x32 bf16 WMMA operand fragment (A row-major [M,K] or B stored as
// W[N,K] row-major == column-major KxN). Per the CDNA5 16-bit operand layout:
// lane<16 holds row r, K = k0+0..7 and k0+16..23 ; lane>=16 holds K +8.
// Two 16-byte loads per lane -> global_load_b128.
__device__ __forceinline__ bf16x16 load_frag(const u16* base, int row0,
                                             long long ld, int k0) {
  int lane = threadIdx.x & 31;
  int half = lane >> 4, r = lane & 15;
  const u16* p = base + (long long)(row0 + r) * ld + k0 + half * 8;
  bf16x16 f;
  ((uint4*)&f)[0] = *(const uint4*)p;        // K = k0+half*8 .. +7
  ((uint4*)&f)[1] = *(const uint4*)(p + 16); // K = k0+16+half*8 .. +7
  return f;
}

// fp32 -> bf16 convert (weights)
__global__ void __launch_bounds__(256) cvt_bf16_kernel(const float* __restrict__ in,
                                                       u16* __restrict__ out,
                                                       long long n) {
  long long i = (long long)blockIdx.x * blockDim.x + threadIdx.x;
  long long stride = (long long)gridDim.x * blockDim.x;
  for (; i < n; i += stride) out[i] = f2bf(in[i]);
}

// LayerNorm: fp32 row [C] -> bf16 row
__global__ void __launch_bounds__(256) ln_bf16_kernel(const float* __restrict__ x,
                                                      const float* __restrict__ g,
                                                      const float* __restrict__ be,
                                                      u16* __restrict__ out) {
  __shared__ float s1[256], s2[256];
  int row = blockIdx.x, t = threadIdx.x;
  const float* xr = x + (size_t)row * CQ;
  float v[4], s = 0.f, sq = 0.f;
#pragma unroll
  for (int i = 0; i < 4; ++i) { v[i] = xr[t + 256*i]; s += v[i]; sq += v[i]*v[i]; }
  s1[t] = s; s2[t] = sq; __syncthreads();
  for (int off = 128; off > 0; off >>= 1) {
    if (t < off) { s1[t] += s1[t+off]; s2[t] += s2[t+off]; }
    __syncthreads();
  }
  float mu  = s1[0] * (1.f/CQ);
  float var = s2[0] * (1.f/CQ) - mu*mu;
  float rin = rsqrtf(var + 1e-5f);
  u16* orow = out + (size_t)row * CQ;
#pragma unroll
  for (int i = 0; i < 4; ++i) {
    int c = t + 256*i;
    orow[c] = f2bf((v[i] - mu) * rin * g[c] + be[c]);
  }
}

// repack v from qkv[(l*B+b), 2C + c] -> vT[b][c][l]  (K-pad cols stay zero)
// grid: (ceil(L/256), C, B) -> writes coalesced along l
__global__ void __launch_bounds__(256) vrepack_kernel(const u16* __restrict__ qkv,
                                                      u16* __restrict__ vT) {
  int l = blockIdx.x * 256 + threadIdx.x;
  if (l >= LQ) return;
  int c = blockIdx.y, b = blockIdx.z;
  vT[((long long)b * CQ + c) * LP + l] =
      qkv[((long long)l * BQ + b) * (3*CQ) + 2*CQ + c];
}

// Attention scores + dual softmax + head-average, per (b, 16-row block).
// attn[b][l][n] = (1/(2H)) * sum_h [ softmax(bias + s*qq^T) + softmax(bias + s*kk^T) ]
__global__ void __launch_bounds__(256) sclip_attention_kernel(
    const u16* __restrict__ qkv, const float* __restrict__ att_bias,
    u16* __restrict__ attn) {
  __shared__ float sc[16][LP];                    // 38 KB
  const int ltile = blockIdx.x;                   // 0..36
  const int b = blockIdx.y;
  const int tid = threadIdx.x;
  const int wave = tid >> 5, lane = tid & 31;
  const int l0 = ltile * 16;
  const int NT = (LQ + 15) / 16;                  // 37 col tiles
  const int rr = tid >> 4, jj = tid & 15;         // softmax row / lane-in-row
  const float scale = 0.125f;                     // d^-0.5, d=64

  float accl[LP/16];                              // per-thread attn accumulators
#pragma unroll
  for (int i = 0; i < LP/16; ++i) accl[i] = 0.f;

  for (int h = 0; h < HQ; ++h) {
    const float* bias = att_bias + ((size_t)b * HQ + h) * (size_t)LQ * LQ;
    for (int t = 0; t < 2; ++t) {                 // t=0: q, t=1: k
      __syncthreads();                            // scores free to overwrite
      const u16* base = qkv + (size_t)b * 3*CQ + (size_t)t * CQ + h * DQ;
      for (int nt = wave; nt < NT; nt += 8) {
        int n0 = nt * 16;
        f32x8 cacc = {};
#pragma unroll
        for (int kk = 0; kk < 2; ++kk) {          // K = 64 = 2x32
          bf16x16 a  = load_frag(base, l0, (long long)BQ*3*CQ, kk*32);
          bf16x16 bm = load_frag(base, n0, (long long)BQ*3*CQ, kk*32);
          cacc = __builtin_amdgcn_wmma_f32_16x16x32_bf16(
                     false, a, false, bm, (short)0, cacc, false, false);
        }
        int half = lane >> 4, col = lane & 15;
        int n = n0 + col;
        bool nok = (n < LQ);
        int ncl = nok ? n : (LQ - 1);
        // 8 unconditional, clamped (always in-bounds) bias loads -> one clause
        float bv[8];
#pragma unroll
        for (int r = 0; r < 8; ++r) {
          int l = l0 + r + half*8;
          int lcl = (l < LQ) ? l : (LQ - 1);
          bv[r] = bias[(size_t)lcl * LQ + ncl];
        }
#pragma unroll
        for (int r = 0; r < 8; ++r) {
          int m = r + half*8;
          bool ok = nok && (l0 + m < LQ);
          sc[m][n] = ok ? bv[r] + scale * cacc[r] : -1e30f;
        }
      }
      __syncthreads();                            // scores complete
      // row softmax: 16 lanes per row, shfl_xor width-16 reductions
      float mx = -1e30f;
      for (int n = jj; n < LQ; n += 16) mx = fmaxf(mx, sc[rr][n]);
#pragma unroll
      for (int off = 8; off > 0; off >>= 1) mx = fmaxf(mx, __shfl_xor(mx, off, 16));
      float sum = 0.f;
      for (int n = jj; n < LQ; n += 16) {
        float e = __expf(sc[rr][n] - mx);
        sc[rr][n] = e; sum += e;
      }
#pragma unroll
      for (int off = 8; off > 0; off >>= 1) sum += __shfl_xor(sum, off, 16);
      float inv = 1.f / (sum + 1e-20f);
#pragma unroll
      for (int idx = 0; idx < LP/16; ++idx) {
        int n = jj + 16*idx;
        if (n < LQ) accl[idx] += sc[rr][n] * inv;
      }
    }
  }
  int l = l0 + rr;
  if (l < LQ) {
    u16* orow = attn + ((size_t)b * ATTR + l) * LP;
#pragma unroll
    for (int idx = 0; idx < LP/16; ++idx)
      orow[jj + 16*idx] = f2bf(accl[idx] * (0.5f / HQ));
  }
}

// Generic bf16 WMMA GEMM: out = A[M,K] * W[N,K]^T (+bias)(+QuickGELU)(+res)
// Block: 256 threads = 8 waves; wave -> 32x32 tile; block -> 64(M) x 128(N).
// Software-pipelined K loop: next fragments load while current WMMAs execute.
__global__ void __launch_bounds__(256) gemm_bf16_kernel(
    const u16* __restrict__ A, long long lda, long long sAb,
    const u16* __restrict__ Bm, long long ldb, long long sBb,
    const float* __restrict__ bias, const float* __restrict__ res,
    float* __restrict__ outF, u16* __restrict__ outB,
    long long ldc, long long sOb, int M, int K, int epi) {
  int tid = threadIdx.x, lane = tid & 31, wave = tid >> 5;
  int m0 = blockIdx.y * 64  + (wave >> 2) * 32;
  int n0 = blockIdx.x * 128 + (wave & 3) * 32;
  const u16* Ab = A  + (long long)blockIdx.z * sAb;
  const u16* Bb = Bm + (long long)blockIdx.z * sBb;
  long long ob  = (long long)blockIdx.z * sOb;

  f32x8 c00 = {}, c01 = {}, c10 = {}, c11 = {};
  bf16x16 a0 = load_frag(Ab, m0,      lda, 0);
  bf16x16 a1 = load_frag(Ab, m0 + 16, lda, 0);
  bf16x16 b0 = load_frag(Bb, n0,      ldb, 0);
  bf16x16 b1 = load_frag(Bb, n0 + 16, ldb, 0);
  for (int k0 = 32; k0 < K; k0 += 32) {
    bf16x16 na0 = load_frag(Ab, m0,      lda, k0);
    bf16x16 na1 = load_frag(Ab, m0 + 16, lda, k0);
    bf16x16 nb0 = load_frag(Bb, n0,      ldb, k0);
    bf16x16 nb1 = load_frag(Bb, n0 + 16, ldb, k0);
    c00 = __builtin_amdgcn_wmma_f32_16x16x32_bf16(false, a0, false, b0, (short)0, c00, false, false);
    c01 = __builtin_amdgcn_wmma_f32_16x16x32_bf16(false, a0, false, b1, (short)0, c01, false, false);
    c10 = __builtin_amdgcn_wmma_f32_16x16x32_bf16(false, a1, false, b0, (short)0, c10, false, false);
    c11 = __builtin_amdgcn_wmma_f32_16x16x32_bf16(false, a1, false, b1, (short)0, c11, false, false);
    a0 = na0; a1 = na1; b0 = nb0; b1 = nb1;
  }
  c00 = __builtin_amdgcn_wmma_f32_16x16x32_bf16(false, a0, false, b0, (short)0, c00, false, false);
  c01 = __builtin_amdgcn_wmma_f32_16x16x32_bf16(false, a0, false, b1, (short)0, c01, false, false);
  c10 = __builtin_amdgcn_wmma_f32_16x16x32_bf16(false, a1, false, b0, (short)0, c10, false, false);
  c11 = __builtin_amdgcn_wmma_f32_16x16x32_bf16(false, a1, false, b1, (short)0, c11, false, false);

  int half = lane >> 4, col = lane & 15;
  f32x8 cc[2][2] = {{c00, c01}, {c10, c11}};
#pragma unroll
  for (int mt = 0; mt < 2; ++mt) {
#pragma unroll
    for (int nt = 0; nt < 2; ++nt) {
#pragma unroll
      for (int r = 0; r < 8; ++r) {
        int m = m0 + mt*16 + r + half*8;
        int n = n0 + nt*16 + col;
        if (m < M) {
          float v = cc[mt][nt][r];
          if (bias) v += bias[n];
          if (epi)  v *= 1.f / (1.f + __expf(-1.702f * v));   // QuickGELU
          long long off = ob + (long long)m * ldc + n;
          if (res)  v += res[off];
          if (outF) outF[off] = v;
          if (outB) outB[off] = f2bf(v);
        }
      }
    }
  }
}

// ---------------------------------------------------------------------------
extern "C" void kernel_launch(void* const* d_in, const int* in_sizes, int n_in,
                              void* d_out, int out_size, void* d_ws, size_t ws_size,
                              hipStream_t stream) {
  (void)in_sizes; (void)n_in; (void)out_size;
  const float* x_in  = (const float*)d_in[0];
  const float* attb  = (const float*)d_in[1];
  const float* ln1_g = (const float*)d_in[2];
  const float* ln1_b = (const float*)d_in[3];
  const float* in_w  = (const float*)d_in[4];
  const float* in_b  = (const float*)d_in[5];
  const float* out_w = (const float*)d_in[6];
  const float* out_b = (const float*)d_in[7];
  const float* ln2_g = (const float*)d_in[8];
  const float* ln2_b = (const float*)d_in[9];
  const float* fc1_w = (const float*)d_in[10];
  const float* fc1_b = (const float*)d_in[11];
  const float* fc2_w = (const float*)d_in[12];
  const float* fc2_b = (const float*)d_in[13];
  float* x = (float*)d_out;

  // workspace carve-up (bf16 buffers, 256B aligned)
  char* ws = (char*)d_ws;
  size_t off = 0;
  auto alloc = [&](size_t elems) {
    u16* p = (u16*)(ws + off);
    off += ((elems * 2 + 255) & ~(size_t)255);
    return p;
  };
  u16* wInB  = alloc((size_t)3*CQ*CQ);     // per-layer bf16 weights
  u16* wOutB = alloc((size_t)CQ*CQ);
  u16* wFc1B = alloc((size_t)DFF*CQ);
  u16* wFc2B = alloc((size_t)CQ*DFF);
  u16* hB    = alloc((size_t)LBP*CQ);      // LN output (M-padded, zero pad)
  u16* qkvB  = alloc((size_t)LBP*3*CQ);
  u16* vTB   = alloc((size_t)BQ*CQ*LP);    // v transposed, K-padded
  u16* attnB = alloc((size_t)BQ*ATTR*LP);  // averaged attention, M/K-padded
  u16* oaB   = alloc((size_t)LBP*CQ);      // attn @ v
  u16* uB    = alloc((size_t)LBP*DFF);     // gelu(fc1)
  size_t total = off;
  if (total > ws_size) return;

  hipMemsetAsync(d_ws, 0, total, stream);  // zero pads every call (ws poisoned)
  hipMemcpyAsync(x, x_in, (size_t)LB*CQ*sizeof(float),
                 hipMemcpyDeviceToDevice, stream);

  for (int i = 0; i < NLAY; ++i) {
    // per-layer weight conversion fp32 -> bf16
    cvt_bf16_kernel<<<2048, 256, 0, stream>>>(in_w  + (size_t)i*3*CQ*CQ, wInB,  (long long)3*CQ*CQ);
    cvt_bf16_kernel<<<2048, 256, 0, stream>>>(out_w + (size_t)i*CQ*CQ,   wOutB, (long long)CQ*CQ);
    cvt_bf16_kernel<<<2048, 256, 0, stream>>>(fc1_w + (size_t)i*DFF*CQ,  wFc1B, (long long)DFF*CQ);
    cvt_bf16_kernel<<<2048, 256, 0, stream>>>(fc2_w + (size_t)i*CQ*DFF,  wFc2B, (long long)CQ*DFF);

    // LN1
    ln_bf16_kernel<<<LB, 256, 0, stream>>>(x, ln1_g + i*CQ, ln1_b + i*CQ, hB);

    // qkv = h * in_w^T + in_b   [LB, 3C] bf16
    gemm_bf16_kernel<<<dim3(3*CQ/128, (LB+63)/64, 1), 256, 0, stream>>>(
        hB, CQ, 0, wInB, CQ, 0, in_b + (size_t)i*3*CQ,
        nullptr, nullptr, qkvB, 3*CQ, 0, LB, CQ, 0);

    // vT[b][c][l]
    vrepack_kernel<<<dim3((LQ+255)/256, CQ, BQ), 256, 0, stream>>>(qkvB, vTB);

    // attn[b][l][n]
    sclip_attention_kernel<<<dim3((LQ+15)/16, BQ), 256, 0, stream>>>(
        qkvB, attb + (size_t)i*BQ*HQ*LQ*LQ, attnB);

    // out_attn[(l,b),c] = attn[b] @ v[b]   (batched over b)
    gemm_bf16_kernel<<<dim3(CQ/128, (LQ+63)/64, BQ), 256, 0, stream>>>(
        attnB, LP, (long long)ATTR*LP, vTB, LP, (long long)CQ*LP,
        nullptr, nullptr, nullptr, oaB, (long long)BQ*CQ, CQ, LQ, LP, 0);

    // x += out_attn * out_w^T + out_b
    gemm_bf16_kernel<<<dim3(CQ/128, (LB+63)/64, 1), 256, 0, stream>>>(
        oaB, CQ, 0, wOutB, CQ, 0, out_b + (size_t)i*CQ,
        x, x, nullptr, CQ, 0, LB, CQ, 0);

    // LN2
    ln_bf16_kernel<<<LB, 256, 0, stream>>>(x, ln2_g + i*CQ, ln2_b + i*CQ, hB);

    // u = quickgelu(h2 * fc1_w^T + fc1_b)   [LB, DFF] bf16
    gemm_bf16_kernel<<<dim3(DFF/128, (LB+63)/64, 1), 256, 0, stream>>>(
        hB, CQ, 0, wFc1B, CQ, 0, fc1_b + (size_t)i*DFF,
        nullptr, nullptr, uB, DFF, 0, LB, CQ, 1);

    // x += u * fc2_w^T + fc2_b
    gemm_bf16_kernel<<<dim3(CQ/128, (LB+63)/64, 1), 256, 0, stream>>>(
        uB, DFF, 0, wFc2B, DFF, 0, fc2_b + (size_t)i*CQ,
        x, x, nullptr, CQ, 0, LB, DFF, 0);
  }
}
